// DecoderBlock_78692390797843
// MI455X (gfx1250) — matrix operand order repaired and checked by reading, not compiled, not run on hardware
//
#include <hip/hip_runtime.h>

typedef __bf16 bf16;
typedef __attribute__((ext_vector_type(16))) __bf16 v16bf;
typedef __attribute__((ext_vector_type(8)))  __bf16 v8bf;
typedef __attribute__((ext_vector_type(8)))  float  v8f;

#define DIM    2048
#define NHEADS 16
#define HD     128
#define HIDDEN 8192
#define BSZ    2
#define SEQ    2048
#define ROWS   (BSZ*SEQ)

// ---------------------------------------------------------------- helpers

__device__ __forceinline__ v8f wmma_bf16(v16bf a, v16bf b, v8f c) {
  // D = A(16x32 bf16) x B(32x16 bf16) + C(16x16 f32)
  return __builtin_amdgcn_wmma_f32_16x16x32_bf16(false, a, false, b,
                                                 (short)0, c, false, false);
}

// A-fragment (16x32, 16-bit): lane m holds K = hl*16..hl*16+15 contiguous
__device__ __forceinline__ v16bf afrag(const bf16* p, int hl) {
  union { v16bf v; v8bf h[2]; } u;
  u.h[0] = *(const v8bf*)(p + hl * 16);
  u.h[1] = *(const v8bf*)(p + hl * 16 + 8);
  return u.v;
}

// B-fragment (32x16, 16-bit): lane n holds column n, K = hl*16..hl*16+15
// contiguous; loaded as two 16B halves (needs only 16B alignment).
__device__ __forceinline__ v16bf bfrag(const bf16* p, int hl) {
  union { v16bf v; v8bf h[2]; } u;
  u.h[0] = *(const v8bf*)(p + hl * 16);
  u.h[1] = *(const v8bf*)(p + hl * 16 + 8);
  return u.v;
}

__device__ __forceinline__ v8f vzero8() {
  v8f z;
#pragma unroll
  for (int i = 0; i < 8; ++i) z[i] = 0.f;
  return z;
}

// CDNA5 async global->LDS copy (16B per lane), tracked by ASYNCcnt.
__device__ __forceinline__ void async_copy_b128(const bf16* gptr, bf16* lptr) {
  unsigned lds_off = (unsigned)(uintptr_t)lptr;  // low 32 bits of the shared pointer are the LDS offset
  asm volatile("global_load_async_to_lds_b128 %0, %1, off"
               :: "v"(lds_off), "v"(gptr)
               : "memory");
}

template <int N>
__device__ __forceinline__ void async_wait() {
  asm volatile("s_wait_asynccnt %0" :: "i"(N) : "memory");
}

// ---------------------------------------------------------------- f32 -> bf16

__global__ void cvt_bf16_kernel(const float* __restrict__ in,
                                bf16* __restrict__ out, int n) {
  int i = blockIdx.x * blockDim.x + threadIdx.x;
  int stride = gridDim.x * blockDim.x;
  for (; i < n; i += stride) out[i] = (bf16)in[i];
}

// ---------------------------------------------------------------- RMSNorm

__global__ __launch_bounds__(256)
void rmsnorm_kernel(const float* __restrict__ x, const float* __restrict__ w,
                    bf16* __restrict__ out, int ncols) {
  __shared__ float red[8];
  int row = blockIdx.x;
  const float* xr = x + (size_t)row * ncols;
  float ss = 0.f;
  for (int c = threadIdx.x; c < ncols; c += 256) {
    float v = xr[c];
    ss += v * v;
  }
#pragma unroll
  for (int m = 16; m >= 1; m >>= 1) ss += __shfl_xor(ss, m, 32);
  if ((threadIdx.x & 31) == 0) red[threadIdx.x >> 5] = ss;
  __syncthreads();
  if (threadIdx.x == 0) {
    float tot = 0.f;
#pragma unroll
    for (int i = 0; i < 8; ++i) tot += red[i];
    red[0] = rsqrtf(tot / (float)ncols + 1e-6f);
  }
  __syncthreads();
  float scale = red[0];
  for (int c = threadIdx.x; c < ncols; c += 256)
    out[(size_t)row * ncols + c] = (bf16)(xr[c] * scale * w[c]);
}

// ---------------------------------------------------------------- WMMA GEMM
// C[M,N] = A[M,K](bf16) x W[N,K](bf16)^T  (+ epilogue)
// MODE 0: bf16 out = acc * alpha ; MODE 1: bf16 relu ; MODE 2: f32 + res
//
// Block tile 128x256, 8 waves in 2(M) x 4(N); each wave computes 64x64.
// Double-buffered LDS fed by async global->LDS copies (software pipeline):
// issue group kt+1, s_wait_asynccnt 6 (group kt done, in-order), barrier,
// compute 16 WMMAs from buffer kt.

#define GTM 128
#define GTN 256
#define GTK 32
#define GLD 40   // LDS row stride (elements): 80B rows -> conflict-free 16B lanes

template <int MODE>
__global__ __launch_bounds__(256)
void gemm_wmma_kernel(const bf16* __restrict__ A, const bf16* __restrict__ W,
                      void* __restrict__ out, const float* __restrict__ res,
                      int M, int N, int K, float alpha) {
  __shared__ bf16 As[2][GTM * GLD];   // 2 x 10240B
  __shared__ bf16 Ws[2][GTN * GLD];   // 2 x 20480B

  int n0 = blockIdx.x * GTN, m0 = blockIdx.y * GTM;
  int t = threadIdx.x;
  int wave = t >> 5, lane = t & 31;
  int wm = wave >> 2, wn = wave & 3;       // 2 x 4 wave grid
  int hl = lane >> 4, ln = lane & 15;

  v8f acc[4][4];
#pragma unroll
  for (int i = 0; i < 4; ++i)
#pragma unroll
    for (int j = 0; j < 4; ++j) acc[i][j] = vzero8();

  // async stage one 32-wide K slice into buffer `buf`
  auto issue = [&](int buf, int k0) {
#pragma unroll
    for (int i = 0; i < 2; ++i) {
      int c = t + i * 256;
      int row = c >> 2, col = (c & 3) * 8;
      async_copy_b128(A + (size_t)(m0 + row) * K + k0 + col,
                      As[buf] + row * GLD + col);
    }
#pragma unroll
    for (int i = 0; i < 4; ++i) {
      int c = t + i * 256;
      int row = c >> 2, col = (c & 3) * 8;
      async_copy_b128(W + (size_t)(n0 + row) * K + k0 + col,
                      Ws[buf] + row * GLD + col);
    }
  };

  int nk = K / GTK;
  issue(0, 0);
  for (int kt = 0; kt < nk; ++kt) {
    int buf = kt & 1;
    if (kt + 1 < nk) {
      issue(buf ^ 1, (kt + 1) * GTK);
      async_wait<6>();   // group kt complete; group kt+1 still in flight
    } else {
      async_wait<0>();
    }
    __syncthreads();

    v16bf af[4], bfv[4];
#pragma unroll
    for (int i = 0; i < 4; ++i)
      af[i] = afrag(As[buf] + (wm * 64 + i * 16 + ln) * GLD, hl);
#pragma unroll
    for (int j = 0; j < 4; ++j)
      bfv[j] = bfrag(Ws[buf] + (wn * 64 + j * 16 + ln) * GLD, hl);

#pragma unroll
    for (int i = 0; i < 4; ++i)
#pragma unroll
      for (int j = 0; j < 4; ++j)
        acc[i][j] = wmma_bf16(af[i], bfv[j], acc[i][j]);
    __syncthreads();   // readers done before buffer `buf` is refilled
  }

  // epilogue: C/D layout: VGPR r -> row r + 8*hl, lane -> col ln
#pragma unroll
  for (int i = 0; i < 4; ++i)
#pragma unroll
    for (int j = 0; j < 4; ++j) {
      int col = n0 + wn * 64 + j * 16 + ln;
#pragma unroll
      for (int r = 0; r < 8; ++r) {
        int row = m0 + wm * 64 + i * 16 + r + 8 * hl;
        size_t idx = (size_t)row * N + col;
        float v = acc[i][j][r];
        if (MODE == 2)
          ((float*)out)[idx] = v + res[idx];
        else if (MODE == 1)
          ((bf16*)out)[idx] = (bf16)(v > 0.f ? v : 0.f);
        else
          ((bf16*)out)[idx] = (bf16)(v * alpha);
      }
    }
}

// ---------------------------------------------------------------- attention
// q,k,v: bf16 [B,S,DIM] (head-major inner dim); q pre-scaled by 1/sqrt(HD).
// One wave handles 16 query rows; flash-style streaming over 32-key blocks.
// V tile is copied async to LDS while QK^T + softmax run.

#define VLD 144  // LDS stride for V tile rows (elements)
#define PLD 48

__global__ __launch_bounds__(128)
void attn_kernel(const bf16* __restrict__ q, const bf16* __restrict__ k,
                 const bf16* __restrict__ v, bf16* __restrict__ out) {
  __shared__ bf16 Vs[4][32 * VLD];
  __shared__ bf16 Ps[4][16 * PLD];

  int b = blockIdx.z, h = blockIdx.y;
  int wave = threadIdx.x >> 5, lane = threadIdx.x & 31;
  int hl = lane >> 4, ln = lane & 15;
  int q0 = blockIdx.x * 64 + wave * 16;

  const bf16* qb = q + (size_t)b * SEQ * DIM + h * HD;
  const bf16* kb = k + (size_t)b * SEQ * DIM + h * HD;
  const bf16* vb = v + (size_t)b * SEQ * DIM + h * HD;
  bf16* Vw = Vs[wave];
  bf16* Pw = Ps[wave];

  // Q fragments: 4 k-chunks of 32 over HD=128, held in registers
  v16bf qf[4];
  {
    const bf16* qr = qb + (size_t)(q0 + ln) * DIM;
#pragma unroll
    for (int c = 0; c < 4; ++c) qf[c] = afrag(qr + c * 32, hl);
  }

  v8f o[8];
#pragma unroll
  for (int d = 0; d < 8; ++d) o[d] = vzero8();
  float mrow[8], lrow[8];
#pragma unroll
  for (int r = 0; r < 8; ++r) { mrow[r] = -1e30f; lrow[r] = 0.f; }

  int nkb = (q0 >> 5) + 1;  // causal: key blocks covering keys <= q0+15
  for (int kbi = 0; kbi < nkb; ++kbi) {
    int kbase = kbi * 32;

    // async-stage V tile [32 x 128] into this wave's LDS region
    // (overlaps with the QK^T WMMAs and the softmax below)
#pragma unroll
    for (int i = 0; i < 16; ++i) {
      int c = i * 32 + lane;
      int r = c >> 4, col = (c & 15) * 8;
      async_copy_b128(vb + (size_t)(kbase + r) * DIM + col,
                      Vw + r * VLD + col);
    }

    // preload all K fragments, then run the 8 score WMMAs back-to-back
    v16bf kf[2][4];
#pragma unroll
    for (int nt = 0; nt < 2; ++nt) {
      const bf16* kcol = kb + (size_t)(kbase + nt * 16 + ln) * DIM;
#pragma unroll
      for (int c = 0; c < 4; ++c) kf[nt][c] = bfrag(kcol + c * 32, hl);
    }
    v8f st[2];
#pragma unroll
    for (int nt = 0; nt < 2; ++nt) {
      v8f s = vzero8();
#pragma unroll
      for (int c = 0; c < 4; ++c) s = wmma_bf16(qf[c], kf[nt][c], s);
      st[nt] = s;
    }

    // online softmax (per-row stats replicated across the 16-lane half)
    float corr[8];
#pragma unroll
    for (int r = 0; r < 8; ++r) {
      int row = q0 + r + 8 * hl;
      float s0 = st[0][r], s1 = st[1][r];
      if (kbase + ln > row) s0 = -1e30f;
      if (kbase + 16 + ln > row) s1 = -1e30f;
      float bm = fmaxf(s0, s1);
#pragma unroll
      for (int m = 8; m >= 1; m >>= 1) bm = fmaxf(bm, __shfl_xor(bm, m, 32));
      float mn = fmaxf(mrow[r], bm);
      float c0 = __expf(mrow[r] - mn);
      float p0 = __expf(s0 - mn), p1 = __expf(s1 - mn);
      float rs = p0 + p1;
#pragma unroll
      for (int m = 8; m >= 1; m >>= 1) rs += __shfl_xor(rs, m, 32);
      lrow[r] = lrow[r] * c0 + rs;
      mrow[r] = mn;
      corr[r] = c0;
      Pw[(r + 8 * hl) * PLD + ln] = (bf16)p0;
      Pw[(r + 8 * hl) * PLD + 16 + ln] = (bf16)p1;
    }
#pragma unroll
    for (int d = 0; d < 8; ++d)
#pragma unroll
      for (int r = 0; r < 8; ++r) o[d][r] *= corr[r];

    // O += P(16x32) x V(32x128): P restaged via LDS into A-layout
    v16bf pf = afrag(Pw + ln * PLD, hl);
    async_wait<0>();   // V tile resident in LDS (per-wave region)
#pragma unroll
    for (int d = 0; d < 8; ++d) {
      v16bf vf;
#pragma unroll
      for (int e = 0; e < 16; ++e)
        vf[e] = Vw[(hl * 16 + e) * VLD + d * 16 + ln];
      o[d] = wmma_bf16(pf, vf, o[d]);
    }
  }

  // normalize and store bf16
#pragma unroll
  for (int d = 0; d < 8; ++d)
#pragma unroll
    for (int r = 0; r < 8; ++r) {
      int row = q0 + r + 8 * hl;
      out[((size_t)b * SEQ + row) * DIM + h * HD + d * 16 + ln] =
          (bf16)(o[d][r] / lrow[r]);
    }
}

// ---------------------------------------------------------------- launch

extern "C" void kernel_launch(void* const* d_in, const int* in_sizes, int n_in,
                              void* d_out, int out_size, void* d_ws, size_t ws_size,
                              hipStream_t stream) {
  (void)in_sizes; (void)n_in; (void)out_size; (void)ws_size;
  const float* x   = (const float*)d_in[0];
  // d_in[1] = mask (causal, recomputed analytically)
  const float* wq  = (const float*)d_in[2];
  const float* wk  = (const float*)d_in[3];
  const float* wv  = (const float*)d_in[4];
  const float* wo  = (const float*)d_in[5];
  const float* w1  = (const float*)d_in[6];
  const float* w2  = (const float*)d_in[7];
  const float* anw = (const float*)d_in[8];
  const float* fnw = (const float*)d_in[9];

  char* ws = (char*)d_ws;
  size_t off = 0;
  auto alloc = [&](size_t bytes) { void* p = ws + off; off += bytes; return p; };

  bf16* WQb = (bf16*)alloc((size_t)DIM * DIM * 2);
  bf16* WKb = (bf16*)alloc((size_t)DIM * DIM * 2);
  bf16* WVb = (bf16*)alloc((size_t)DIM * DIM * 2);
  bf16* WOb = (bf16*)alloc((size_t)DIM * DIM * 2);
  bf16* W1b = (bf16*)alloc((size_t)HIDDEN * DIM * 2);
  bf16* W2b = (bf16*)alloc((size_t)DIM * HIDDEN * 2);
  bf16* XB  = (bf16*)alloc((size_t)ROWS * DIM * 2);
  bf16* Qb  = (bf16*)alloc((size_t)ROWS * DIM * 2);
  bf16* Kb  = (bf16*)alloc((size_t)ROWS * DIM * 2);
  bf16* Vb  = (bf16*)alloc((size_t)ROWS * DIM * 2);
  bf16* AT  = (bf16*)alloc((size_t)ROWS * DIM * 2);
  float* Hf = (float*)alloc((size_t)ROWS * DIM * 4);
  bf16* HN  = (bf16*)alloc((size_t)ROWS * DIM * 2);
  bf16* MID = (bf16*)alloc((size_t)ROWS * HIDDEN * 2);

  // weights -> bf16
  cvt_bf16_kernel<<<1024, 256, 0, stream>>>(wq, WQb, DIM * DIM);
  cvt_bf16_kernel<<<1024, 256, 0, stream>>>(wk, WKb, DIM * DIM);
  cvt_bf16_kernel<<<1024, 256, 0, stream>>>(wv, WVb, DIM * DIM);
  cvt_bf16_kernel<<<1024, 256, 0, stream>>>(wo, WOb, DIM * DIM);
  cvt_bf16_kernel<<<2048, 256, 0, stream>>>(w1, W1b, HIDDEN * DIM);
  cvt_bf16_kernel<<<2048, 256, 0, stream>>>(w2, W2b, DIM * HIDDEN);

  // attn rmsnorm
  rmsnorm_kernel<<<ROWS, 256, 0, stream>>>(x, anw, XB, DIM);

  // QKV projections (softmax scale folded into Q)
  dim3 gProj(DIM / GTN, ROWS / GTM);      // (8, 32)
  const float qscale = 0.08838834764831845f;  // 1/sqrt(128)
  gemm_wmma_kernel<0><<<gProj, 256, 0, stream>>>(XB, WQb, Qb, nullptr,
                                                 ROWS, DIM, DIM, qscale);
  gemm_wmma_kernel<0><<<gProj, 256, 0, stream>>>(XB, WKb, Kb, nullptr,
                                                 ROWS, DIM, DIM, 1.0f);
  gemm_wmma_kernel<0><<<gProj, 256, 0, stream>>>(XB, WVb, Vb, nullptr,
                                                 ROWS, DIM, DIM, 1.0f);

  // causal attention
  attn_kernel<<<dim3(SEQ / 64, NHEADS, BSZ), 128, 0, stream>>>(Qb, Kb, Vb, AT);

  // output projection + residual -> h (f32)
  gemm_wmma_kernel<2><<<gProj, 256, 0, stream>>>(AT, WOb, Hf, x,
                                                 ROWS, DIM, DIM, 1.0f);

  // ffn rmsnorm
  rmsnorm_kernel<<<ROWS, 256, 0, stream>>>(Hf, fnw, HN, DIM);

  // FFN up + relu
  dim3 gUp(HIDDEN / GTN, ROWS / GTM);     // (32, 32)
  gemm_wmma_kernel<1><<<gUp, 256, 0, stream>>>(HN, W1b, MID, nullptr,
                                               ROWS, HIDDEN, DIM, 1.0f);

  // FFN down + residual -> out (f32)
  gemm_wmma_kernel<2><<<gProj, 256, 0, stream>>>(MID, W2b, d_out, Hf,
                                                 ROWS, DIM, HIDDEN, 1.0f);
}
